// CrystalGraphConvNet_88940182766274
// MI455X (gfx1250) — matrix operand order repaired and checked by reading, not compiled, not run on hardware
//
#include <hip/hip_runtime.h>

typedef __attribute__((ext_vector_type(16))) _Float16 v16h;
typedef __attribute__((ext_vector_type(8)))  _Float16 v8h;
typedef __attribute__((ext_vector_type(8)))  float    v8f;

#define NATOM 100000
#define NPAD  100096       // 782 * 128, row-padded x so GEMM stores need no guards
#define MNBR  12
#define ORIGF 92
#define NBRF  41
#define FDIM  64
#define HDIM  128
#define BCRY  128
#define K2F   128          // 2F output channels of conv GEMM
#define KTOT  169          // 2F + NBR_F
#define KPAD  192          // padded to 6 * 32
#define KEMB  96           // 92 padded to 3 * 32
#define NMROWS (NATOM * MNBR)
#define EPSBN 1e-5f

__device__ __forceinline__ float sp_f(float x) {          // stable softplus
    return fmaxf(x, 0.0f) + log1pf(expf(-fabsf(x)));
}
__device__ __forceinline__ float sig_f(float x) {
    return 1.0f / (1.0f + expf(-x));
}

// pack 4 floats -> 4 f16, one 8-byte LDS store
__device__ __forceinline__ void st4h(_Float16* p, float4 v) {
    union { _Float16 h[4]; unsigned long long u; } q;
    q.h[0] = (_Float16)v.x; q.h[1] = (_Float16)v.y;
    q.h[2] = (_Float16)v.z; q.h[3] = (_Float16)v.w;
    *(unsigned long long*)p = q.u;
}

// A fragment: 16x32 f16 rows from LDS tile (row-major, stride in halves)
__device__ __forceinline__ v16h load_a_frag(const _Float16* base, int lane, int stride, int k0) {
    int m    = lane & 15;
    int ksel = (lane >> 4) * 8;
    const _Float16* p = base + m * stride + k0 + ksel;
    v8h lo = *(const v8h*)(p);
    v8h hi = *(const v8h*)(p + 16);
    v16h f;
#pragma unroll
    for (int i = 0; i < 8; ++i) { f[i] = lo[i]; f[8 + i] = hi[i]; }
    return f;
}

// B fragment: 32x16 f16; weights pre-transposed as [outC][Kpad] row-major
__device__ __forceinline__ v16h load_b_frag(const _Float16* wt, int lane, int kstride, int colbase, int k0) {
    int c = colbase + (lane & 15);
    const _Float16* p = wt + c * kstride + k0 + (lane >> 4) * 8;
    v8h lo = *(const v8h*)(p);
    v8h hi = *(const v8h*)(p + 16);
    v16h f;
#pragma unroll
    for (int i = 0; i < 8; ++i) { f[i] = lo[i]; f[8 + i] = hi[i]; }
    return f;
}

__global__ void zero_kernel(float* p, int n) {
    int i = blockIdx.x * blockDim.x + threadIdx.x;
    if (i < n) p[i] = 0.0f;
}

// Transpose + pad + f16-convert weights: W_emb [92,64]->[64][96], W_full_l [169,128]->[128][192]
__global__ void prep_weights(const float* Wemb, const float* Wf0, const float* Wf1, const float* Wf2,
                             _Float16* wtEmb, _Float16* wtFull) {
    const int embN = FDIM * KEMB;          // 6144
    const int fulN = K2F * KPAD;           // 24576
    int total = embN + 3 * fulN;
    for (int i = blockIdx.x * blockDim.x + threadIdx.x; i < total; i += gridDim.x * blockDim.x) {
        if (i < embN) {
            int c = i / KEMB, k = i % KEMB;
            wtEmb[i] = (k < ORIGF) ? (_Float16)Wemb[k * FDIM + c] : (_Float16)0.0f;
        } else {
            int j = i - embN;
            int l = j / fulN, r = j % fulN;
            int c = r / KPAD, k = r % KPAD;
            const float* W = (l == 0) ? Wf0 : (l == 1) ? Wf1 : Wf2;
            wtFull[l * fulN + r] = (k < KTOT) ? (_Float16)W[k * K2F + c] : (_Float16)0.0f;
        }
    }
}

// ---- Embedding GEMM: x = atom_fea @ W_emb + b  ([N,92]x[92,64]) via f16 WMMA ----
__global__ __launch_bounds__(256) void embed_kernel(const float* __restrict__ af,
                                                    const _Float16* __restrict__ wt,
                                                    const float* __restrict__ bias,
                                                    float* __restrict__ x) {
    __shared__ __align__(16) _Float16 At[128 * KEMB];
    int t = threadIdx.x;
    int rowblock = blockIdx.x * 128;
    // fill A tile: 128 rows x 96 halves, 2 threads/row (48 halves each)
    {
        int r    = t >> 1;
        int half = t & 1;
        int a    = rowblock + r;
        _Float16* dst = At + r * KEMB;
        if (a < NATOM) {
            const float* src = af + (size_t)a * ORIGF;   // 368 B rows: 16B aligned
            if (half == 0) {
#pragma unroll
                for (int c = 0; c < 48; c += 4) st4h(dst + c, *(const float4*)(src + c));
            } else {
#pragma unroll
                for (int c = 48; c < 92; c += 4) {
                    float4 v = *(const float4*)(src + c);
                    if (c == 88) v.w = 0.0f;             // col 91 zeroed, 92..95 padded below
                    st4h(dst + c, v);
                }
                for (int c = 92; c < KEMB; ++c) dst[c] = (_Float16)0.0f;
            }
        } else {
            for (int c = half * 48; c < half * 48 + 48; ++c) dst[c] = (_Float16)0.0f;
        }
    }
    __syncthreads();

    int wave = t >> 5, lane = t & 31;
    const _Float16* Ab = At + wave * 16 * KEMB;
    v16h afr[3];
#pragma unroll
    for (int kk = 0; kk < 3; ++kk) afr[kk] = load_a_frag(Ab, lane, KEMB, kk * 32);

#pragma unroll
    for (int ct = 0; ct < 4; ++ct) {
        int col = ct * 16 + (lane & 15);
        float b = bias[col];
        v8f acc;
#pragma unroll
        for (int v = 0; v < 8; ++v) acc[v] = b;
#pragma unroll
        for (int kk = 0; kk < 3; ++kk) {
            v16h bfr = load_b_frag(wt, lane, KEMB, ct * 16, kk * 32);
            acc = __builtin_amdgcn_wmma_f32_16x16x32_f16(false, afr[kk], false, bfr,
                                                         (short)0, acc, false, false);
        }
        // x padded to NPAD rows -> unconditional stores (no exec-mask branches)
        size_t base = (size_t)(rowblock + wave * 16 + (lane >> 4) * 8) * FDIM + col;
#pragma unroll
        for (int v = 0; v < 8; ++v) x[base + (size_t)v * FDIM] = acc[v];
    }
}

// ---- Conv GEMM pass 1: g = concat(self, gather, nbr) @ W_full + b ; BN1 stats ----
__global__ __launch_bounds__(256) void conv_gemm(const float* __restrict__ x,
                                                 const float* __restrict__ nbr,
                                                 const int* __restrict__ idx,
                                                 const _Float16* __restrict__ wt,
                                                 const float* __restrict__ bias,
                                                 _Float16* __restrict__ g,
                                                 float* __restrict__ stats) {
    __shared__ __align__(16) _Float16 At[128 * KPAD];   // 48 KB
    __shared__ float shS[K2F], shSS[K2F];
    int t = threadIdx.x;
    if (t < K2F) { shS[t] = 0.0f; shSS[t] = 0.0f; }
    int rowblock = blockIdx.x * 128;
    // fill A tile: row = a*12+m ; cols [0,64)=x[a], [64,128)=x[nb], [128,169)=nbr, pad 0
    {
        int r    = t >> 1;
        int half = t & 1;
        int grow = rowblock + r;                         // NMROWS divisible by 128
        int a = grow / MNBR, m = grow % MNBR;
        int nb = idx[a * MNBR + m];
        const float* xa = x + (size_t)a * FDIM;          // 256 B aligned rows
        const float* xn = x + (size_t)nb * FDIM;
        _Float16* dst = At + r * KPAD;
        if (half == 0) {
#pragma unroll
            for (int c = 0; c < 64; c += 4) st4h(dst + c, *(const float4*)(xa + c));
#pragma unroll
            for (int c = 0; c < 32; c += 4) st4h(dst + 64 + c, *(const float4*)(xn + c));
        } else {
#pragma unroll
            for (int c = 32; c < 64; c += 4) st4h(dst + 64 + c, *(const float4*)(xn + c));
            const float* nf = nbr + ((size_t)a * MNBR + m) * NBRF;
            for (int c = 0; c < NBRF; ++c) dst[128 + c] = (_Float16)nf[c];
            for (int c = KTOT; c < KPAD; ++c) dst[c] = (_Float16)0.0f;
        }
    }
    __syncthreads();

    int wave = t >> 5, lane = t & 31;
    const _Float16* Ab = At + wave * 16 * KPAD;
    v16h afr[6];
#pragma unroll
    for (int kk = 0; kk < 6; ++kk) afr[kk] = load_a_frag(Ab, lane, KPAD, kk * 32);

#pragma unroll 1
    for (int ct = 0; ct < 8; ++ct) {
        int col = ct * 16 + (lane & 15);
        float b = bias[col];
        v8f acc;
#pragma unroll
        for (int v = 0; v < 8; ++v) acc[v] = b;
#pragma unroll
        for (int kk = 0; kk < 6; ++kk) {
            v16h bfr = load_b_frag(wt, lane, KPAD, ct * 16, kk * 32);
            acc = __builtin_amdgcn_wmma_f32_16x16x32_f16(false, afr[kk], false, bfr,
                                                         (short)0, acc, false, false);
        }
        size_t base = (size_t)(rowblock + wave * 16 + (lane >> 4) * 8) * K2F + col;
        float s = 0.0f, ss = 0.0f;
#pragma unroll
        for (int v = 0; v < 8; ++v) {
            float d = acc[v];
            g[base + (size_t)v * K2F] = (_Float16)d;
            s += d; ss += d * d;
        }
        atomicAdd(&shS[col], s);
        atomicAdd(&shSS[col], ss);
    }
    __syncthreads();
    if (t < K2F) {
        atomicAdd(&stats[t], shS[t]);
        atomicAdd(&stats[K2F + t], shSS[t]);
    }
}

// fold BN stats + gamma/beta into per-channel scale/shift
__global__ void finalize_stats(const float* stats, const float* gamma, const float* beta,
                               float* scale, float* shift, int C, float invcnt) {
    int c = threadIdx.x;
    if (c < C) {
        float mean = stats[c] * invcnt;
        float var  = stats[C + c] * invcnt - mean * mean;
        float sc   = gamma[c] * rsqrtf(var + EPSBN);
        scale[c] = sc;
        shift[c] = beta[c] - mean * sc;
    }
}

// ---- pass 2: BN1 + sigmoid(filt)*softplus(core), sum over M ; BN2 stats ----
__global__ __launch_bounds__(256) void conv_reduce(const _Float16* __restrict__ g,
                                                   const float* __restrict__ scale1,
                                                   const float* __restrict__ shift1,
                                                   float* __restrict__ summed,
                                                   float* __restrict__ stats2) {
    __shared__ float sS[FDIM], sSS[FDIM];
    int t = threadIdx.x;
    if (t < FDIM) { sS[t] = 0.0f; sSS[t] = 0.0f; }
    __syncthreads();
    int c = t & 63;
    int a = blockIdx.x * 4 + (t >> 6);
    if (a < NATOM) {
        float scF = scale1[c],        shF = shift1[c];
        float scC = scale1[FDIM + c], shC = shift1[FDIM + c];
        const _Float16* gp = g + (size_t)a * MNBR * K2F;
        float acc = 0.0f;
#pragma unroll
        for (int m = 0; m < MNBR; ++m) {
            float f  = (float)gp[m * K2F + c]        * scF + shF;
            float co = (float)gp[m * K2F + FDIM + c] * scC + shC;
            acc += sig_f(f) * sp_f(co);
        }
        summed[(size_t)a * FDIM + c] = acc;
        atomicAdd(&sS[c], acc);
        atomicAdd(&sSS[c], acc * acc);
    }
    __syncthreads();
    if (t < FDIM) {
        atomicAdd(&stats2[t], sS[t]);
        atomicAdd(&stats2[FDIM + t], sSS[t]);
    }
}

// x = softplus(x + bn2(summed)), in place
__global__ void update_x(float* __restrict__ x, const float* __restrict__ summed,
                         const float* __restrict__ scale2, const float* __restrict__ shift2) {
    int i = blockIdx.x * blockDim.x + threadIdx.x;
    if (i < NATOM * FDIM) {
        int c = i & 63;
        x[i] = sp_f(x[i] + summed[i] * scale2[c] + shift2[c]);
    }
}

// segment pooling via searchsorted(crystal_ends, a, 'right') + atomics
__global__ void pool_kernel(const float* __restrict__ x, const int* __restrict__ ce,
                            float* __restrict__ crysSum, float* __restrict__ cnt) {
    int i = blockIdx.x * blockDim.x + threadIdx.x;
    if (i < NATOM * FDIM) {
        int a = i >> 6, c = i & 63;
        int lo = 0, hi = BCRY;
        while (lo < hi) { int mid = (lo + hi) >> 1; if (ce[mid] <= a) lo = mid + 1; else hi = mid; }
        atomicAdd(&crysSum[lo * FDIM + c], x[i]);
        if (c == 0) atomicAdd(&cnt[lo], 1.0f);
    }
}

// head: out[b] = softplus(softplus(crys) @ W_fc + b_fc) @ W_out + b_out
__global__ __launch_bounds__(128) void head_kernel(const float* __restrict__ crysSum,
                                                   const float* __restrict__ cnt,
                                                   const float* __restrict__ Wfc,
                                                   const float* __restrict__ bfc,
                                                   const float* __restrict__ Wout,
                                                   const float* __restrict__ bout,
                                                   float* __restrict__ out) {
    __shared__ float sh[HDIM];
    int b = blockIdx.x, j = threadIdx.x;
    float inv = 1.0f / fmaxf(cnt[b], 1.0f);
    float hj = bfc[j];
#pragma unroll 4
    for (int k = 0; k < FDIM; ++k)
        hj += sp_f(crysSum[b * FDIM + k] * inv) * Wfc[k * HDIM + j];
    sh[j] = sp_f(hj) * Wout[j];
    __syncthreads();
    for (int s = 64; s > 0; s >>= 1) {
        if (j < s) sh[j] += sh[j + s];
        __syncthreads();
    }
    if (j == 0) out[b] = sh[0] + bout[0];
}

extern "C" void kernel_launch(void* const* d_in, const int* in_sizes, int n_in,
                              void* d_out, int out_size, void* d_ws, size_t ws_size,
                              hipStream_t stream) {
    // ---- inputs (setup_inputs dict order, params flattened in literal order) ----
    const float* atom_fea = (const float*)d_in[0];
    const float* nbr_fea  = (const float*)d_in[1];
    const int*   nbr_idx  = (const int*)d_in[2];
    const int*   ce       = (const int*)d_in[3];
    const float* W_emb    = (const float*)d_in[4];
    const float* b_emb    = (const float*)d_in[5];
    const float* Wf[3],  *bf[3], *g1[3], *be1[3], *g2[3], *be2[3];
    for (int l = 0; l < 3; ++l) {
        Wf[l]  = (const float*)d_in[6 + 6 * l];
        bf[l]  = (const float*)d_in[7 + 6 * l];
        g1[l]  = (const float*)d_in[8 + 6 * l];
        be1[l] = (const float*)d_in[9 + 6 * l];
        g2[l]  = (const float*)d_in[10 + 6 * l];
        be2[l] = (const float*)d_in[11 + 6 * l];
    }
    const float* W_fc  = (const float*)d_in[24];
    const float* b_fc  = (const float*)d_in[25];
    const float* W_out = (const float*)d_in[26];
    const float* b_out = (const float*)d_in[27];
    float* out = (float*)d_out;

    // ---- workspace carve-up (all offsets 16B aligned) ----
    char* ws = (char*)d_ws;
    float*    x       = (float*)ws;                              ws += (size_t)NPAD * FDIM * 4;    // 25.6 MB (padded)
    float*    summed  = (float*)ws;                              ws += (size_t)NATOM * FDIM * 4;   // 25.6 MB
    _Float16* g       = (_Float16*)ws;                           ws += (size_t)NMROWS * K2F * 2;   // 307 MB
    _Float16* wtEmb   = (_Float16*)ws;                           ws += (size_t)FDIM * KEMB * 2;
    _Float16* wtFull  = (_Float16*)ws;                           ws += (size_t)3 * K2F * KPAD * 2;
    float*    stats1  = (float*)ws;                              ws += 256 * 4;
    float*    stats2  = (float*)ws;                              ws += 128 * 4;
    float*    scale1  = (float*)ws;                              ws += 128 * 4;
    float*    shift1  = (float*)ws;                              ws += 128 * 4;
    float*    scale2  = (float*)ws;                              ws += 64 * 4;
    float*    shift2  = (float*)ws;                              ws += 64 * 4;
    float*    crysSum = (float*)ws;                              ws += BCRY * FDIM * 4;
    float*    cnt     = (float*)ws;                              ws += BCRY * 4;

    // ---- pipeline ----
    prep_weights<<<312, 256, 0, stream>>>(W_emb, Wf[0], Wf[1], Wf[2], wtEmb, wtFull);
    embed_kernel<<<NPAD / 128, 256, 0, stream>>>(atom_fea, wtEmb, b_emb, x);

    for (int l = 0; l < 3; ++l) {
        zero_kernel<<<2, 256, 0, stream>>>(stats1, 256 + 128);   // stats1 + stats2 are adjacent
        conv_gemm<<<NMROWS / 128, 256, 0, stream>>>(x, nbr_fea, nbr_idx,
                                                    wtFull + (size_t)l * K2F * KPAD,
                                                    bf[l], g, stats1);
        finalize_stats<<<1, 128, 0, stream>>>(stats1, g1[l], be1[l], scale1, shift1,
                                              K2F, 1.0f / (float)NMROWS);
        conv_reduce<<<(NATOM + 3) / 4, 256, 0, stream>>>(g, scale1, shift1, summed, stats2);
        finalize_stats<<<1, 64, 0, stream>>>(stats2, g2[l], be2[l], scale2, shift2,
                                             FDIM, 1.0f / (float)NATOM);
        update_x<<<(NATOM * FDIM + 255) / 256, 256, 0, stream>>>(x, summed, scale2, shift2);
    }

    zero_kernel<<<(BCRY * FDIM + BCRY + 255) / 256, 256, 0, stream>>>(crysSum, BCRY * FDIM + BCRY);
    pool_kernel<<<(NATOM * FDIM + 255) / 256, 256, 0, stream>>>(x, ce, crysSum, cnt);
    head_kernel<<<BCRY, 128, 0, stream>>>(crysSum, cnt, W_fc, b_fc, W_out, b_out, out);
}
